// GCN_80401787781186
// MI455X (gfx1250) — matrix operand order repaired and checked by reading, not compiled, not run on hardware
//
#include <hip/hip_runtime.h>
#include <hip/hip_bf16.h>

typedef __attribute__((ext_vector_type(16))) __bf16 v16bf;
typedef __attribute__((ext_vector_type(8)))  __bf16 v8bf;
typedef __attribute__((ext_vector_type(8)))  float  v8f;

#define F_DIM 128
#define N_GRAPHS 2048
#define LDS_STRIDE 136   // 128 + 8 bf16 pad -> 4-bank advance per row, conflict-free b128 reads

__device__ __forceinline__ float selu_f(float x) {
    const float l = 1.0507009873554805f;
    const float la = 1.7580993408473766f; // lambda*alpha
    return x > 0.f ? l * x : la * (__expf(x) - 1.f);
}

// ---------------- degree / norm ----------------
__global__ void k_deg(const int* __restrict__ dst, float* __restrict__ deg, int E) {
    int e = blockIdx.x * blockDim.x + threadIdx.x;
    if (e < E) atomicAdd(&deg[dst[e]], 1.0f);
}

__global__ void k_dis(float* __restrict__ deg, int N) {
    int i = blockIdx.x * blockDim.x + threadIdx.x;
    if (i < N) deg[i] = rsqrtf(deg[i] + 1.0f);   // +1 self-loop; always > 0
}

__global__ void k_norm(const int* __restrict__ src, const int* __restrict__ dst,
                       const float* __restrict__ dis, float* __restrict__ norm, int E) {
    int e = blockIdx.x * blockDim.x + threadIdx.x;
    if (e < E) norm[e] = dis[src[e]] * dis[dst[e]];
}

// ---------------- conversions ----------------
__global__ void k_tobf16(const float* __restrict__ s, __bf16* __restrict__ d, int n) {
    int i = blockIdx.x * blockDim.x + threadIdx.x;
    if (i < n) d[i] = (__bf16)s[i];
}

// Wt[o][i] = W[i][o], f32 -> bf16 (so WMMA B-fragments read contiguously)
__global__ void k_wtrans(const float* __restrict__ W, __bf16* __restrict__ Wt, int In, int Out) {
    int k = blockIdx.x * blockDim.x + threadIdx.x;
    if (k >= In * Out) return;
    int i = k / Out, o = k % Out;
    Wt[(size_t)o * In + i] = (__bf16)W[(size_t)i * Out + o];
}

// ---------------- WMMA GEMM: C[M x 128] = A[M x 128] @ W[128 x 128] ----------------
// A bf16 row-major, Wt bf16 = W transposed (Wt[n][k]). Wt staged in LDS per block.
// block: 256 (8 waves); wave w owns row tile blockIdx.x*8 + w, computes full 16x128 strip.
// 8 accumulators; all 4 A-fragments preloaded; B-fragments software-pipelined 1 ahead.
// mode 0: store lin only
// mode 1: store lin AND agg = lin * dis[row]^2 + bias[col]   (GCN self-loop + bias init)
// mode 2: store selu(lin + bias[col])                         (MLP layer)
__global__ void k_gemm_bf16(const __bf16* __restrict__ A, const __bf16* __restrict__ Wt,
                            const float* __restrict__ bias, const float* __restrict__ dis,
                            float* __restrict__ C, float* __restrict__ agg,
                            int M, int mode) {
    __shared__ __bf16 sW[F_DIM * LDS_STRIDE];

    // ---- cooperative stage of Wt into LDS (32KB, padded rows) ----
    {
        int t = threadIdx.x;                 // 0..255
        int r = t >> 1, h = t & 1;           // row, half
        const v8bf* sp = (const v8bf*)(Wt + (size_t)r * F_DIM + h * 64);
        v8bf*       dp = (v8bf*)(sW + (size_t)r * LDS_STRIDE + h * 64);
#pragma unroll
        for (int i = 0; i < 8; ++i) dp[i] = sp[i];
    }
    __syncthreads();

    const int wave = threadIdx.x >> 5;
    const int lane = threadIdx.x & 31;
    const int tile = blockIdx.x * 8 + wave;
    if (tile >= (M >> 4)) return;            // wave-uniform exit (EXEC stays full for WMMA)

    const int row0 = tile * 16;
    const int m  = lane & 15;                // A row within tile
    const int kb = (lane >> 4) * 16;         // A: hi half-lanes take K+16
    const int ko = (lane >> 4) * 16;         // B: hi half-lanes take K+16
    const int n  = lane & 15;                // B column within tile
    const __bf16* arow = A + (size_t)(row0 + m) * F_DIM;
    const __bf16* wbase = sW + (size_t)n * LDS_STRIDE + ko;

    // preload all 4 A fragments (full 16x128 strip, 32 VGPRs)
    v16bf afrag[4];
#pragma unroll
    for (int kk = 0; kk < 4; ++kk) {
        v8bf a_lo = *(const v8bf*)(arow + kk * 32 + kb);
        v8bf a_hi = *(const v8bf*)(arow + kk * 32 + 8 + kb);
#pragma unroll
        for (int i = 0; i < 8; ++i) { afrag[kk][i] = a_lo[i]; afrag[kk][i + 8] = a_hi[i]; }
    }

    v8f acc[8] = {};
    // flattened (k0, ct) loop, B fragment pipelined one iteration ahead
    v8bf bl0 = *(const v8bf*)(wbase);
    v8bf bh0 = *(const v8bf*)(wbase + 8);
#pragma unroll
    for (int it = 0; it < 32; ++it) {
        v8bf bl1, bh1;
        if (it < 31) {
            const int itn = it + 1;
            const __bf16* p = wbase + (size_t)((itn & 7) * 16) * LDS_STRIDE + (itn >> 3) * 32;
            bl1 = *(const v8bf*)(p);
            bh1 = *(const v8bf*)(p + 8);
        }
        v16bf b;
#pragma unroll
        for (int i = 0; i < 8; ++i) { b[i] = bl0[i]; b[i + 8] = bh0[i]; }
        const int ct = it & 7, kk = it >> 3;
        acc[ct] = __builtin_amdgcn_wmma_f32_16x16x32_bf16(false, afrag[kk], false, b,
                                                          (short)0, acc[ct], false, false);
        bl0 = bl1; bh0 = bh1;
    }

    // C/D layout: lanes 0-15 -> N=lane, M=v; lanes 16-31 -> N=lane-16, M=8+v
    const int rbase = row0 + (lane >> 4) * 8;
#pragma unroll
    for (int ct = 0; ct < 8; ++ct) {
        const int col = ct * 16 + n;
        const float bv = bias ? bias[col] : 0.f;
#pragma unroll
        for (int v = 0; v < 8; ++v) {
            const int row = rbase + v;
            float x = acc[ct][v];
            size_t idx = (size_t)row * F_DIM + col;
            if (mode == 2) {
                C[idx] = selu_f(x + bv);
            } else {
                C[idx] = x;
                if (mode == 1) {
                    float dd = dis[row];
                    agg[idx] = x * dd * dd + bv;
                }
            }
        }
    }
}

// ---------------- edge scatter-add: agg[dst] += lin[src] * norm ----------------
// persistent waves; lane handles 4 consecutive features; prefetch next gather row
__global__ void k_agg(const int* __restrict__ src, const int* __restrict__ dst,
                      const float* __restrict__ norm, const float* __restrict__ lin,
                      float* __restrict__ agg, int E) {
    int lane = threadIdx.x & 31;
    int wid  = (blockIdx.x * blockDim.x + threadIdx.x) >> 5;
    int nw   = (gridDim.x * blockDim.x) >> 5;
    for (int e = wid; e < E; e += nw) {
        int en = e + nw;
        if (en < E) {
            int sp = src[en];
            __builtin_prefetch(lin + (size_t)sp * F_DIM + lane * 4, 0, 0); // covers 512B row
        }
        int s = src[e], d = dst[e];
        float nm = norm[e];
        float4 v = ((const float4*)(lin + (size_t)s * F_DIM))[lane];
        float* ad = agg + (size_t)d * F_DIM + lane * 4;
        atomicAdd(ad + 0, v.x * nm);
        atomicAdd(ad + 1, v.y * nm);
        atomicAdd(ad + 2, v.z * nm);
        atomicAdd(ad + 3, v.w * nm);
    }
}

// ---------------- BN statistics over selu(agg) (stats only, no writeback) ----------------
#define ROWS_PER_BLK 256
__global__ void k_finalize_bnstat(const float* __restrict__ agg, float* __restrict__ csum,
                                  float* __restrict__ csum2, int N) {
    int f  = threadIdx.x;
    int r0 = blockIdx.x * ROWS_PER_BLK;
    float s = 0.f, s2 = 0.f;
    for (int r = 0; r < ROWS_PER_BLK; ++r) {
        int i = r0 + r;
        if (i >= N) break;
        float v = selu_f(agg[(size_t)i * F_DIM + f]);
        s += v; s2 += v * v;
    }
    atomicAdd(&csum[f], s);
    atomicAdd(&csum2[f], s2);
}

// ---------------- fold BN stats into per-feature scale/shift ----------------
__global__ void k_bn_prep(const float* __restrict__ csum, const float* __restrict__ csum2,
                          const float* __restrict__ gamma, const float* __restrict__ beta,
                          float* __restrict__ scale, float* __restrict__ shift, int N) {
    int f = threadIdx.x;
    float invn = 1.0f / (float)N;
    float mean = csum[f] * invn;
    float var  = csum2[f] * invn - mean * mean;
    float sc   = gamma[f] * rsqrtf(var + 1e-5f);
    scale[f] = sc;
    shift[f] = beta[f] - mean * sc;
}

// ---------------- batchnorm apply (recompute selu), emit bf16 for next GEMM ----------------
__global__ void k_bn_apply(const float* __restrict__ agg, const float* __restrict__ scale,
                           const float* __restrict__ shift, __bf16* __restrict__ outbf, int N) {
    int idx = blockIdx.x * blockDim.x + threadIdx.x;
    if (idx >= N * F_DIM) return;
    int f = idx & (F_DIM - 1);
    float v = selu_f(agg[idx]);
    outbf[idx] = (__bf16)(v * scale[f] + shift[f]);
}

// ---------------- layer 3 finalize: ReLU + atomic global_add_pool ----------------
__global__ void k_finalize_pool(const float* __restrict__ agg, const int* __restrict__ batch,
                                float* __restrict__ hg, int N) {
    int f = threadIdx.x;
    int i = blockIdx.x;
    if (i >= N) return;
    float v = fmaxf(agg[(size_t)i * F_DIM + f], 0.f);
    atomicAdd(&hg[(size_t)batch[i] * F_DIM + f], v);
}

// ---------------- final projection [128 -> 1] ----------------
__global__ void k_final_dot(const float* __restrict__ hm, const float* __restrict__ fw2,
                            const float* __restrict__ fb2, float* __restrict__ out, int G) {
    int wid  = (blockIdx.x * blockDim.x + threadIdx.x) >> 5;
    int lane = threadIdx.x & 31;
    if (wid >= G) return;
    float4 a = ((const float4*)(hm + (size_t)wid * F_DIM))[lane];
    float4 b = ((const float4*)fw2)[lane];
    float s = a.x * b.x + a.y * b.y + a.z * b.z + a.w * b.w;
#pragma unroll
    for (int off = 16; off > 0; off >>= 1) s += __shfl_down(s, off, 32);
    if (lane == 0) out[wid] = s + fb2[0];
}

// =======================================================================
extern "C" void kernel_launch(void* const* d_in, const int* in_sizes, int n_in,
                              void* d_out, int out_size, void* d_ws, size_t ws_size,
                              hipStream_t stream) {
    const float* x    = (const float*)d_in[0];
    const int*   ei   = (const int*)d_in[1];
    const int*   batch= (const int*)d_in[2];
    const float* W1   = (const float*)d_in[3];
    const float* b1   = (const float*)d_in[4];
    const float* g1   = (const float*)d_in[5];
    const float* be1  = (const float*)d_in[6];
    const float* W2   = (const float*)d_in[7];
    const float* b2   = (const float*)d_in[8];
    const float* g2   = (const float*)d_in[9];
    const float* be2  = (const float*)d_in[10];
    const float* W3   = (const float*)d_in[11];
    const float* b3   = (const float*)d_in[12];
    const float* fw1  = (const float*)d_in[13];
    const float* fb1  = (const float*)d_in[14];
    const float* fw2  = (const float*)d_in[15];
    const float* fb2  = (const float*)d_in[16];

    const int N = in_sizes[0] / F_DIM;       // 100000
    const int E = in_sizes[1] / 2;           // 1600000
    const int* src = ei;
    const int* dst = ei + E;

    // ---- workspace carve-up (256B aligned) ----
    char* ws = (char*)d_ws;
    size_t off = 0;
    auto carve = [&](size_t bytes) { char* p = ws + off; off += (bytes + 255) & ~(size_t)255; return p; };
    float*  dis   = (float*)carve((size_t)N * 4);
    float*  norm  = (float*)carve((size_t)E * 4);
    float*  lin   = (float*)carve((size_t)N * F_DIM * 4);
    float*  agg   = (float*)carve((size_t)N * F_DIM * 4);
    __bf16* hbf   = (__bf16*)carve((size_t)N * F_DIM * 2);
    __bf16* wt    = (__bf16*)carve((size_t)F_DIM * F_DIM * 2);
    float*  csum  = (float*)carve(F_DIM * 4);
    float*  csum2 = (float*)carve(F_DIM * 4);
    float*  scale = (float*)carve(F_DIM * 4);
    float*  shift = (float*)carve(F_DIM * 4);
    float*  hg    = (float*)carve((size_t)N_GRAPHS * F_DIM * 4);
    __bf16* hgbf  = (__bf16*)carve((size_t)N_GRAPHS * F_DIM * 2);
    float*  hm    = (float*)carve((size_t)N_GRAPHS * F_DIM * 4);

    const int T = 256;
    const int nfeat = N * F_DIM;

    // ---- degree / normalization coefficients ----
    hipMemsetAsync(dis, 0, (size_t)N * 4, stream);
    k_deg <<<(E + T - 1) / T, T, 0, stream>>>(dst, dis, E);
    k_dis <<<(N + T - 1) / T, T, 0, stream>>>(dis, N);
    k_norm<<<(E + T - 1) / T, T, 0, stream>>>(src, dst, dis, norm, E);

    // ---- x -> bf16 ----
    k_tobf16<<<(nfeat + T - 1) / T, T, 0, stream>>>(x, hbf, nfeat);

    const int ntiles   = N / 16;                         // 6250
    const int gemmGrid = (ntiles + 7) / 8;
    const int aggGrid  = 2048;                           // persistent waves
    const int fbBlocks = (N + ROWS_PER_BLK - 1) / ROWS_PER_BLK;

    const float* Ws[2]  = {W1, W2};
    const float* bs[2]  = {b1, b2};
    const float* gs[2]  = {g1, g2};
    const float* bes[2] = {be1, be2};

    // ---- GCN layers 1 & 2: GEMM(+self-loop init) -> scatter -> SELU stats -> BN ----
    for (int L = 0; L < 2; ++L) {
        k_wtrans<<<(F_DIM * F_DIM + T - 1) / T, T, 0, stream>>>(Ws[L], wt, F_DIM, F_DIM);
        k_gemm_bf16<<<gemmGrid, T, 0, stream>>>(hbf, wt, bs[L], dis, lin, agg, N, 1);
        k_agg<<<aggGrid, T, 0, stream>>>(src, dst, norm, lin, agg, E);
        hipMemsetAsync(csum, 0, F_DIM * 4, stream);
        hipMemsetAsync(csum2, 0, F_DIM * 4, stream);
        k_finalize_bnstat<<<fbBlocks, F_DIM, 0, stream>>>(agg, csum, csum2, N);
        k_bn_prep<<<1, F_DIM, 0, stream>>>(csum, csum2, gs[L], bes[L], scale, shift, N);
        k_bn_apply<<<(nfeat + T - 1) / T, T, 0, stream>>>(agg, scale, shift, hbf, N);
    }

    // ---- GCN layer 3: GEMM(+self-loop init) -> scatter -> ReLU -> pooled ----
    k_wtrans<<<(F_DIM * F_DIM + T - 1) / T, T, 0, stream>>>(W3, wt, F_DIM, F_DIM);
    k_gemm_bf16<<<gemmGrid, T, 0, stream>>>(hbf, wt, b3, dis, lin, agg, N, 1);
    k_agg<<<aggGrid, T, 0, stream>>>(src, dst, norm, lin, agg, E);
    hipMemsetAsync(hg, 0, (size_t)N_GRAPHS * F_DIM * 4, stream);
    k_finalize_pool<<<N, F_DIM, 0, stream>>>(agg, batch, hg, N);

    // ---- MLP head: selu(hg @ fw1 + fb1) @ fw2 + fb2 ----
    k_tobf16<<<(N_GRAPHS * F_DIM + T - 1) / T, T, 0, stream>>>(hg, hgbf, N_GRAPHS * F_DIM);
    k_wtrans<<<(F_DIM * F_DIM + T - 1) / T, T, 0, stream>>>(fw1, wt, F_DIM, F_DIM);
    k_gemm_bf16<<<(N_GRAPHS / 16 + 7) / 8, T, 0, stream>>>(hgbf, wt, fb1, nullptr, hm, nullptr,
                                                           N_GRAPHS, 2);
    k_final_dot<<<(N_GRAPHS * 32 + T - 1) / T, T, 0, stream>>>(hm, fw2, fb2, (float*)d_out, N_GRAPHS);
}
